// BinaryDecorator_8907762172383
// MI455X (gfx1250) — 8-bit matrix operand order repaired and checked by reading, compiled, not run on hardware
//
#include <hip/hip_runtime.h>

typedef __attribute__((ext_vector_type(8))) int v8i;

#define TOKENS 16384
#define D_IN   4096
#define D_OUT  4096


__global__ void init_ws(float* __restrict__ sum) {
    if (threadIdx.x == 0 && blockIdx.x == 0) *sum = 0.0f;
}

__device__ __forceinline__ signed char sgn8(float v) {
    return (signed char)((v > 0.0f) - (v < 0.0f));
}

__global__ __launch_bounds__(256) void quantize_x(const float* __restrict__ x,
                                                  signed char* __restrict__ sx,
                                                  float* __restrict__ sum) {
    size_t t = (size_t)blockIdx.x * blockDim.x + threadIdx.x;
    const float4* xv = (const float4*)x + t * 4;
    float local = 0.0f;
    __attribute__((aligned(16))) signed char q[16];
#pragma unroll
    for (int i = 0; i < 4; ++i) {
        float4 v = xv[i];
        local += fabsf(v.x) + fabsf(v.y) + fabsf(v.z) + fabsf(v.w);
        q[4 * i + 0] = sgn8(v.x);
        q[4 * i + 1] = sgn8(v.y);
        q[4 * i + 2] = sgn8(v.z);
        q[4 * i + 3] = sgn8(v.w);
    }
    ((int4*)sx)[t] = *(const int4*)q;
#pragma unroll
    for (int off = 16; off > 0; off >>= 1)
        local += __shfl_xor(local, off, 32);
    if ((threadIdx.x & 31) == 0) atomicAdd(sum, local);
}

__global__ __launch_bounds__(256) void quantize_w(const float* __restrict__ w,
                                                  signed char* __restrict__ sw) {
    size_t t = (size_t)blockIdx.x * blockDim.x + threadIdx.x;
    const float4* wv = (const float4*)w + t * 4;
    __attribute__((aligned(16))) signed char q[16];
#pragma unroll
    for (int i = 0; i < 4; ++i) {
        float4 v = wv[i];
        q[4 * i + 0] = sgn8(v.x);
        q[4 * i + 1] = sgn8(v.y);
        q[4 * i + 2] = sgn8(v.z);
        q[4 * i + 3] = sgn8(v.w);
    }
    ((int4*)sw)[t] = *(const int4*)q;
}

__global__ __launch_bounds__(512) void gemm_bin(const signed char* __restrict__ sx,
                                                const signed char* __restrict__ sw,
                                                const float* __restrict__ bias,
                                                const float* __restrict__ sum,
                                                float* __restrict__ out) {
    const int lane = threadIdx.x & 31;
    const int wave = threadIdx.x >> 5;
    const int wm   = wave >> 2;
    const int wn   = wave & 3;
    const int l15  = lane & 15;
    const int hi   = lane >> 4;

    const int m0 = blockIdx.y * 256 + wm * 64;
    const int n0 = blockIdx.x * 256 + wn * 64;

    const signed char* arow0 = sx + (size_t)(m0 + l15) * D_IN + 8 * hi;
    const signed char* brow0 = sw + (size_t)(n0 + l15) * D_IN + 8 * hi;

    v8i acc[16];
#pragma unroll
    for (int i = 0; i < 16; ++i) acc[i] = (v8i)0;

    for (int k0 = 0; k0 < D_IN; k0 += 64) {
        v8i a[4];
#pragma unroll
        for (int mt = 0; mt < 4; ++mt) {
            const signed char* ap = arow0 + (size_t)mt * 16 * D_IN + k0;
            int2 p0 = *(const int2*)(ap +  0);
            int2 p1 = *(const int2*)(ap + 16);
            int2 p2 = *(const int2*)(ap + 32);
            int2 p3 = *(const int2*)(ap + 48);
            a[mt] = (v8i){p0.x, p0.y, p1.x, p1.y, p2.x, p2.y, p3.x, p3.y};
        }

#pragma unroll
        for (int j = 0; j < 4; ++j) {
            const signed char* bp = brow0 + (size_t)j * 16 * D_IN + k0;
            int2 q0 = *(const int2*)(bp +  0);
            int2 q1 = *(const int2*)(bp + 16);
            int2 q2 = *(const int2*)(bp + 32);
            int2 q3 = *(const int2*)(bp + 48);
            v8i bmat = (v8i){q0.x, q0.y, q1.x, q1.y, q2.x, q2.y, q3.x, q3.y};
#pragma unroll
            for (int mt = 0; mt < 4; ++mt) {
                acc[j * 4 + mt] = __builtin_amdgcn_wmma_i32_16x16x64_iu8(
                    true, a[mt], true, bmat, acc[j * 4 + mt], false, false);
            }
        }
    }

    const float scale = *sum * (1.0f / ((float)TOKENS * (float)D_IN));

#pragma unroll
    for (int j = 0; j < 4; ++j) {
        const int n = n0 + j * 16 + l15;
        const float bn = bias[n];
#pragma unroll
        for (int mt = 0; mt < 4; ++mt) {
            const int mb = m0 + mt * 16 + hi * 8;
            const v8i c = acc[j * 4 + mt];
#pragma unroll
            for (int r = 0; r < 8; ++r) {
                out[(size_t)(mb + r) * D_OUT + n] = ((float)c[r] + bn) * scale;
            }
        }
    }
}

extern "C" void kernel_launch(void* const* d_in, const int* in_sizes, int n_in,
                              void* d_out, int out_size, void* d_ws, size_t ws_size,
                              hipStream_t stream) {
    const float* x    = (const float*)d_in[0];
    const float* W    = (const float*)d_in[1];
    const float* bias = (const float*)d_in[2];
    float* out = (float*)d_out;

    char* ws = (char*)d_ws;
    float* sum = (float*)ws;
    signed char* sx = (signed char*)(ws + 256);
    signed char* sw = sx + (size_t)TOKENS * D_IN;

    init_ws<<<1, 1, 0, stream>>>(sum);

    {
        int blocks = (TOKENS * D_IN) / (256 * 16);
        quantize_x<<<blocks, 256, 0, stream>>>(x, sx, sum);
    }
    {
        int blocks = (D_OUT * D_IN) / (256 * 16);
        quantize_w<<<blocks, 256, 0, stream>>>(W, sw);
    }
    {
        dim3 grid(D_OUT / 256, TOKENS / 256);
        gemm_bin<<<grid, 512, 0, stream>>>(sx, sw, bias, sum, out);
    }
}
